// KnowledgeGraphGNN_28509992910888
// MI455X (gfx1250) — matrix operand order repaired and checked by reading, not compiled, not run on hardware
//
#include <hip/hip_runtime.h>
#include <hip/hip_bf16.h>

// ---------------------------------------------------------------------------
// KnowledgeGraphGNN for MI455X (gfx1250, wave32, WMMA)
// B=8, N=1024, C=256, H=512, NL=3, NH=8, dh=64
// ---------------------------------------------------------------------------

typedef __bf16 bf16_t;
typedef __attribute__((ext_vector_type(8)))  bf16_t v8bf;
typedef __attribute__((ext_vector_type(16))) bf16_t v16bf;
typedef __attribute__((ext_vector_type(8)))  float  v8f;

static constexpr int BB   = 8;
static constexpr int NN   = 1024;
static constexpr int CC   = 256;
static constexpr int HH   = 512;
static constexpr int NLAY = 3;
static constexpr int NH_  = 8;
static constexpr int DH_  = 64;

__device__ inline v16bf cat16(v8bf lo, v8bf hi) {
    return __builtin_shufflevector(lo, hi, 0,1,2,3,4,5,6,7,8,9,10,11,12,13,14,15);
}

// ---------------------------------------------------------------------------
// WMMA GEMM: C = A(MxK) * op(B) + bias, bf16 staged in LDS, f32 accumulate.
// BT=true : B given as (Nout x K) row-major  -> C = A * B^T   (weights)
// BT=false: B given as (K x Nout) row-major  -> C = A * B     (adjacency agg)
//
// Block tile 128(M) x 128(N), K-step 32, 256 threads = 8 waves arranged 4x2;
// each wave computes a 32x64 patch as 2x4 v_wmma_f32_16x16x32_bf16 accums
// (8 WMMA per K-step per wave). LDS is double-buffered (2 x 16 KB) so the
// next tile's global loads + bf16 converts overlap the current tile's WMMAs,
// with one workgroup barrier per K-step.
// Requires M%128==0, Nout%128==0, K%32==0 (all call sites satisfy this).
// ---------------------------------------------------------------------------
template<bool BT>
__global__ __launch_bounds__(256)
void wmma_gemm_kernel(const float* __restrict__ A,
                      const float* __restrict__ Bm,
                      const float* __restrict__ bias,
                      float* __restrict__ C,
                      int M, int Nout, int K,
                      long long strideA, long long strideB, long long strideC)
{
    A  += (long long)blockIdx.z * strideA;
    Bm += (long long)blockIdx.z * strideB;
    C  += (long long)blockIdx.z * strideC;

    __shared__ __align__(16) bf16_t As[2][128][32];   // 2 x 8 KB
    __shared__ __align__(16) bf16_t Bs[2][128][32];   // 2 x 8 KB, stored [n][k]

    const int tid  = threadIdx.x;
    const int lane = tid & 31;
    const int wave = tid >> 5;
    const int wm   = wave & 3;       // 0..3 : 32-row slab
    const int wn   = wave >> 2;      // 0..1 : 64-col slab
    const int half = lane >> 4;      // 0/1
    const int lr   = lane & 15;

    const int rowTile = blockIdx.y * 128;
    const int colTile = blockIdx.x * 128;

    // ---- tile staging: global f32 -> bf16 -> LDS ------------------------
    auto stage = [&](int buf, int k0) {
        {   // A tile 128x32: thread handles 16 contiguous k
            const int r    = tid >> 1;
            const int kOff = (tid & 1) * 16;
            const float* ap = A + (long long)(rowTile + r) * K + k0 + kOff;
            if (k0 + 32 < K) __builtin_prefetch(ap + 32, 0, 3);  // global_prefetch_b8
            #pragma unroll
            for (int i = 0; i < 16; ++i)
                As[buf][r][kOff + i] = (bf16_t)ap[i];
        }
        if (BT) {   // B tile from (Nout x K): contiguous in k
            const int n    = tid >> 1;
            const int kOff = (tid & 1) * 16;
            const float* bp = Bm + (long long)(colTile + n) * K + k0 + kOff;
            #pragma unroll
            for (int i = 0; i < 16; ++i)
                Bs[buf][n][kOff + i] = (bf16_t)bp[i];
        } else {    // B tile from (K x Nout): contiguous in n
            const int k    = tid >> 3;
            const int nOff = (tid & 7) * 16;
            const float* bp = Bm + (long long)(k0 + k) * Nout + colTile + nOff;
            #pragma unroll
            for (int i = 0; i < 16; ++i)
                Bs[buf][nOff + i][k] = (bf16_t)bp[i];
        }
    };

    v8f acc[2][4] = {};
    const int KT = K >> 5;

    stage(0, 0);
    int buf = 0;
    for (int kt = 0; kt < KT; ++kt) {
        __syncthreads();                     // buf fully staged / prev reads done
        if (kt + 1 < KT) stage(buf ^ 1, (kt + 1) << 5);

        // ---- fragments per ISA 7.12.2 layouts --------------------------
        v16bf af[2], bfv[4];
        #pragma unroll
        for (int ti = 0; ti < 2; ++ti) {
            // A 16x32 bf16: lane(0-15)=row, K = {0..7,16..23} + 8*half
            const v8bf* ar = (const v8bf*)&As[buf][wm * 32 + ti * 16 + lr][0];
            af[ti] = cat16(ar[half], ar[2 + half]);
        }
        #pragma unroll
        for (int tj = 0; tj < 4; ++tj) {
            // B 32x16 bf16: lane(0-15)=col, K = {0..7,16..23} + 8*half
            const v8bf* br = (const v8bf*)&Bs[buf][wn * 64 + tj * 16 + lr][0];
            bfv[tj] = cat16(br[half], br[2 + half]);
        }
        #pragma unroll
        for (int ti = 0; ti < 2; ++ti)
            #pragma unroll
            for (int tj = 0; tj < 4; ++tj)
                acc[ti][tj] = __builtin_amdgcn_wmma_f32_16x16x32_bf16(
                    false, af[ti], false, bfv[tj], (short)0, acc[ti][tj],
                    false, false);
        buf ^= 1;
    }

    // ---- store: D layout lane->n, vgpr r -> m = 8*half + r
    #pragma unroll
    for (int tj = 0; tj < 4; ++tj) {
        const int gc = colTile + wn * 64 + tj * 16 + lr;
        const float bv = bias ? bias[gc] : 0.0f;
        #pragma unroll
        for (int ti = 0; ti < 2; ++ti) {
            const int gr = rowTile + wm * 32 + ti * 16 + 8 * half;
            #pragma unroll
            for (int r = 0; r < 8; ++r)
                C[(long long)(gr + r) * Nout + gc] = acc[ti][tj][r] + bv;
        }
    }
}

// ---------------------------------------------------------------------------
// Attention over the B=8 "sequence" dim (torch batch_first=False quirk).
// One 64-thread block per (n, head). qkv is (B*N, 3H) row-major.
// ---------------------------------------------------------------------------
__global__ __launch_bounds__(64)
void attn_kernel(const float* __restrict__ qkv, float* __restrict__ att)
{
    const int n = blockIdx.x >> 3;
    const int h = blockIdx.x & 7;
    const int t = threadIdx.x;

    __shared__ float s[8][8];

    // scores s[i][j] = (q_i . k_j) / sqrt(64)
    {
        const int i = t >> 3, j = t & 7;
        const float* qp = qkv + (long long)(i * NN + n) * (3 * HH) + h * DH_;
        const float* kp = qkv + (long long)(j * NN + n) * (3 * HH) + HH + h * DH_;
        float acc = 0.0f;
        #pragma unroll 8
        for (int d = 0; d < DH_; ++d) acc += qp[d] * kp[d];
        s[i][j] = acc * 0.125f;
    }
    __syncthreads();

    // softmax over j (rows handled by threads 0..7)
    if (t < 8) {
        const int i = t;
        float mx = s[i][0];
        #pragma unroll
        for (int j = 1; j < 8; ++j) mx = fmaxf(mx, s[i][j]);
        float sum = 0.0f;
        float e[8];
        #pragma unroll
        for (int j = 0; j < 8; ++j) { e[j] = __expf(s[i][j] - mx); sum += e[j]; }
        const float inv = 1.0f / sum;
        #pragma unroll
        for (int j = 0; j < 8; ++j) s[i][j] = e[j] * inv;
    }
    __syncthreads();

    // o[i, d] = sum_j a[i][j] * v[j][d]
    {
        const int i = t >> 3, dblk = (t & 7) * 8;
        #pragma unroll
        for (int dd = 0; dd < 8; ++dd) {
            const int d = dblk + dd;
            float o = 0.0f;
            #pragma unroll
            for (int j = 0; j < 8; ++j)
                o += s[i][j] * qkv[(long long)(j * NN + n) * (3 * HH) + 2 * HH + h * DH_ + d];
            att[(long long)(i * NN + n) * HH + h * DH_ + d] = o;
        }
    }
}

// ---------------------------------------------------------------------------
// out = LayerNorm(x + agg), no affine, eps 1e-5. One block per 512-elem row.
// ---------------------------------------------------------------------------
__global__ __launch_bounds__(256)
void add_ln_kernel(const float* __restrict__ x, const float* __restrict__ agg,
                   float* __restrict__ out)
{
    const long long base = (long long)blockIdx.x * HH;
    const int t = threadIdx.x;
    __shared__ float rs[256], rs2[256];

    const float e0 = x[base + t]       + agg[base + t];
    const float e1 = x[base + 256 + t] + agg[base + 256 + t];
    rs[t]  = e0 + e1;
    rs2[t] = e0 * e0 + e1 * e1;
    __syncthreads();
    for (int sft = 128; sft > 0; sft >>= 1) {
        if (t < sft) { rs[t] += rs[t + sft]; rs2[t] += rs2[t + sft]; }
        __syncthreads();
    }
    const float mean = rs[0] * (1.0f / HH);
    const float var  = rs2[0] * (1.0f / HH) - mean * mean;
    const float rsq  = rsqrtf(var + 1e-5f);
    out[base + t]       = (e0 - mean) * rsq;
    out[base + 256 + t] = (e1 - mean) * rsq;
}

// ---------------------------------------------------------------------------
// Per-row heads: mastery (H->1), diff (H->5), h1/h2 (W_pre halves).
// One wave per row; wave32 shfl_xor reductions.
// ---------------------------------------------------------------------------
__global__ __launch_bounds__(256)
void heads_kernel(const float* __restrict__ x,
                  const float* __restrict__ Wcls, const float* __restrict__ bcls,
                  const float* __restrict__ Wdiff, const float* __restrict__ bdiff,
                  const float* __restrict__ Wpre,
                  float* __restrict__ mastery, float* __restrict__ diff,
                  float* __restrict__ h1, float* __restrict__ h2)
{
    const int lane = threadIdx.x & 31;
    const int row  = blockIdx.x * 8 + (threadIdx.x >> 5);

    float xv[16];
    #pragma unroll
    for (int k = 0; k < 16; ++k)
        xv[k] = x[(long long)row * HH + lane + 32 * k];

    auto wdot = [&](const float* W) {
        float s = 0.0f;
        #pragma unroll
        for (int k = 0; k < 16; ++k) s += xv[k] * W[lane + 32 * k];
        #pragma unroll
        for (int m = 16; m >= 1; m >>= 1) s += __shfl_xor(s, m, 32);
        return s;
    };

    const float dm  = wdot(Wcls) + bcls[0];
    float dd[5];
    #pragma unroll
    for (int c = 0; c < 5; ++c) dd[c] = wdot(Wdiff + c * HH) + bdiff[c];
    const float p1 = wdot(Wpre);
    const float p2 = wdot(Wpre + HH);

    if (lane == 0) {
        mastery[row] = dm;
        #pragma unroll
        for (int c = 0; c < 5; ++c) diff[row * 5 + c] = dd[c];
        h1[row] = p1;
        h2[row] = p2;
    }
}

// ---------------------------------------------------------------------------
// colmean[b][h] = mean over n of x[b,n,h]
// ---------------------------------------------------------------------------
__global__ __launch_bounds__(256)
void colmean_kernel(const float* __restrict__ x, float* __restrict__ colmean)
{
    const int b = blockIdx.y;
    const int h = blockIdx.x * 256 + threadIdx.x;
    float s = 0.0f;
    const float* p = x + (long long)b * NN * HH + h;
    for (int n = 0; n < NN; ++n) s += p[(long long)n * HH];
    colmean[b * HH + h] = s * (1.0f / NN);
}

// ---------------------------------------------------------------------------
// graph[b] = colmean[b] @ W_pool^T + b_pool  (8 x 512x512 GEMV)
// ---------------------------------------------------------------------------
__global__ __launch_bounds__(256)
void pool_kernel(const float* __restrict__ colmean,
                 const float* __restrict__ Wpool, const float* __restrict__ bpool,
                 float* __restrict__ graph)
{
    const int b = blockIdx.x;
    #pragma unroll
    for (int o = 0; o < 2; ++o) {
        const int ho = threadIdx.x + o * 256;
        float s = bpool[ho];
        const float* mp = colmean + b * HH;
        const float* wp = Wpool + (long long)ho * HH;
        for (int k = 0; k < HH; ++k) s += mp[k] * wp[k];
        graph[b * HH + ho] = s;
    }
}

// ---------------------------------------------------------------------------
// prereq[b, k] = h1[b,i] + h2[b,j] + b_pre, off-diagonal row-major order
// ---------------------------------------------------------------------------
__global__ __launch_bounds__(256)
void prereq_kernel(const float* __restrict__ h1, const float* __restrict__ h2,
                   const float* __restrict__ bpre, float* __restrict__ out)
{
    const int per = NN * (NN - 1);
    const int idx = blockIdx.x * 256 + threadIdx.x;
    const int b  = idx / per;
    const int r  = idx - b * per;
    const int i  = r / (NN - 1);
    const int rr = r - i * (NN - 1);
    const int j  = rr + (rr >= i ? 1 : 0);
    out[idx] = h1[b * NN + i] + h2[b * NN + j] + bpre[0];
}

// ---------------------------------------------------------------------------
// Launch orchestration
// ---------------------------------------------------------------------------
extern "C" void kernel_launch(void* const* d_in, const int* in_sizes, int n_in,
                              void* d_out, int out_size, void* d_ws, size_t ws_size,
                              hipStream_t stream)
{
    (void)in_sizes; (void)n_in; (void)out_size; (void)ws_size;

    const float* cf     = (const float*)d_in[0];   // (B,N,C)
    const float* adj    = (const float*)d_in[1];   // (B,N,N)
    // d_in[2] edge_types, d_in[3] concept_mask: unused (as in reference)
    const float* W_emb  = (const float*)d_in[4];
    const float* b_emb  = (const float*)d_in[5];
    const float* W_in   = (const float*)d_in[6];   // (NL,3H,H)
    const float* b_in   = (const float*)d_in[7];   // (NL,3H)
    const float* W_out  = (const float*)d_in[8];   // (NL,H,H)
    const float* b_out  = (const float*)d_in[9];
    const float* W_conv = (const float*)d_in[10];
    const float* b_conv = (const float*)d_in[11];
    const float* W_cls  = (const float*)d_in[12];
    const float* b_cls  = (const float*)d_in[13];
    const float* W_diff = (const float*)d_in[14];
    const float* b_diff = (const float*)d_in[15];
    const float* W_pre  = (const float*)d_in[16];
    const float* b_pre  = (const float*)d_in[17];
    const float* W_pool = (const float*)d_in[18];
    const float* b_pool = (const float*)d_in[19];

    float* out = (float*)d_out;

    // output layout (floats): x | mastery | diff | prereq | graph
    const long long xOff = 0;
    const long long mOff = xOff + (long long)BB * NN * HH;        // 4,194,304
    const long long dOff = mOff + (long long)BB * NN;             // +8,192
    const long long pOff = dOff + (long long)BB * NN * 5;         // +40,960
    const long long gOff = pOff + (long long)BB * NN * (NN - 1);  // +8,380,416

    // workspace layout (floats)
    float* ws   = (float*)d_ws;
    float* X    = ws;                                  // 8192*512
    float* QKV  = X    + (long long)BB * NN * HH;      // 8192*1536 (also proj out)
    float* ATT  = QKV  + (long long)BB * NN * 3 * HH;  // 8192*512
    float* CONV = ATT  + (long long)BB * NN * HH;      // 8192*512
    float* AGG  = CONV + (long long)BB * NN * HH;      // 8192*512
    float* H1   = AGG  + (long long)BB * NN * HH;      // 8192
    float* H2   = H1   + (long long)BB * NN;           // 8192
    float* CM   = H2   + (long long)BB * NN;           // 8*512
    float* PROJ = QKV;                                 // reuse qkv buffer

    const dim3 blk(256);
    const int M = BB * NN;                             // 8192

    // x = cf @ W_emb^T + b_emb   (8192 x 512 x 256)
    wmma_gemm_kernel<true><<<dim3(HH / 128, M / 128, 1), blk, 0, stream>>>(
        cf, W_emb, b_emb, X, M, HH, CC, 0, 0, 0);

    for (int l = 0; l < NLAY; ++l) {
        const float* Wi = W_in  + (long long)l * 3 * HH * HH;
        const float* bi = b_in  + (long long)l * 3 * HH;
        const float* Wo = W_out + (long long)l * HH * HH;
        const float* bo = b_out + (long long)l * HH;
        const float* Wc = W_conv+ (long long)l * HH * HH;
        const float* bc = b_conv+ (long long)l * HH;

        // qkv = x @ W_in^T + b_in   (8192 x 1536 x 512)
        wmma_gemm_kernel<true><<<dim3(3 * HH / 128, M / 128, 1), blk, 0, stream>>>(
            X, Wi, bi, QKV, M, 3 * HH, HH, 0, 0, 0);

        // attention over B for each (n, head)
        attn_kernel<<<dim3(NN * NH_), dim3(64), 0, stream>>>(QKV, ATT);

        // proj = att @ W_out^T + b_out   (reuses QKV buffer as output)
        wmma_gemm_kernel<true><<<dim3(HH / 128, M / 128, 1), blk, 0, stream>>>(
            ATT, Wo, bo, PROJ, M, HH, HH, 0, 0, 0);

        // conv = proj @ W_conv^T + b_conv
        wmma_gemm_kernel<true><<<dim3(HH / 128, M / 128, 1), blk, 0, stream>>>(
            PROJ, Wc, bc, CONV, M, HH, HH, 0, 0, 0);

        // agg[b] = adj[b] @ conv[b]   (batched: 8 x (1024 x 512 x 1024))
        wmma_gemm_kernel<false><<<dim3(HH / 128, NN / 128, BB), blk, 0, stream>>>(
            adj, CONV, nullptr, AGG, NN, HH, NN,
            (long long)NN * NN, (long long)NN * HH, (long long)NN * HH);

        // x = LN(x + agg); last layer writes straight to output x region
        float* xdst = (l == NLAY - 1) ? (out + xOff) : X;
        add_ln_kernel<<<dim3(M), blk, 0, stream>>>(X, AGG, xdst);
    }

    const float* xf = out + xOff;

    heads_kernel<<<dim3(M / 8), blk, 0, stream>>>(
        xf, W_cls, b_cls, W_diff, b_diff, W_pre,
        out + mOff, out + dOff, H1, H2);

    colmean_kernel<<<dim3(HH / 256, BB), blk, 0, stream>>>(xf, CM);

    pool_kernel<<<dim3(BB), blk, 0, stream>>>(CM, W_pool, b_pool, out + gOff);

    prereq_kernel<<<dim3(BB * NN * (NN - 1) / 256), blk, 0, stream>>>(
        H1, H2, b_pre, out + pOff);
}
